// AttentionHead_59588376264825
// MI455X (gfx1250) — matrix operand order repaired and checked by reading, not compiled, not run on hardware
//
#include <hip/hip_runtime.h>
#include <hip/hip_bf16.h>

typedef __attribute__((ext_vector_type(16))) _Float16 v16h;
typedef __attribute__((ext_vector_type(8)))  _Float16 v8h;
typedef __attribute__((ext_vector_type(8)))  float    v8f;
typedef __attribute__((ext_vector_type(4)))  float    v4f;

#define B_  4
#define T_  4096
#define C_  2048
#define HS_ 128
#define BT_ (B_ * T_)

static __device__ __forceinline__ v8f wmma_f16(v16h a, v16h b, v8f c) {
    return __builtin_amdgcn_wmma_f32_16x16x32_f16(false, a, false, b, (short)0, c, false, false);
}

// ---------------------------------------------------------------------------
// Kernel 0: convert W [C,HS] fp32 (row-major) -> Wt [HS,C] f16 (row-major).
// ---------------------------------------------------------------------------
__global__ __launch_bounds__(256) void wconv_kernel(const float* __restrict__ Wq,
                                                    const float* __restrict__ Wk,
                                                    const float* __restrict__ Wv,
                                                    _Float16* __restrict__ wt) {
    const int which = blockIdx.y;
    const float* W = (which == 0) ? Wq : ((which == 1) ? Wk : Wv);
    int idx = blockIdx.x * 256 + threadIdx.x;      // over C_*HS_
    int k = idx >> 7;
    int n = idx & 127;
    wt[(size_t)which * HS_ * C_ + (size_t)n * C_ + k] = (_Float16)W[idx];
}

// ---------------------------------------------------------------------------
// Kernel 1: QKV projection. One wave computes a 16x128 tile of q, k, or v.
// grid = (BT/16, 3). which: 0 -> qh [BT,HS], 1 -> kh [BT,HS], 2 -> vt [B,HS,T].
// ---------------------------------------------------------------------------
__global__ __launch_bounds__(32) void qkv_kernel(const float* __restrict__ x,
                                                 const _Float16* __restrict__ wt,
                                                 _Float16* __restrict__ qh,
                                                 _Float16* __restrict__ kh,
                                                 _Float16* __restrict__ vt) {
    const int m0    = blockIdx.x * 16;
    const int which = blockIdx.y;
    const _Float16* W = wt + (size_t)which * HS_ * C_;   // [HS][C]
    const int lane = threadIdx.x & 31;
    const int half = lane >> 4;
    const int l16  = lane & 15;

    v8f acc[8];
#pragma unroll
    for (int i = 0; i < 8; ++i) acc[i] = (v8f)(0.0f);

    const float* xrow = x + (size_t)(m0 + l16) * C_;

    for (int k0 = 0; k0 < C_; k0 += 32) {
        // B fragments first: independent load clause, max outstanding.
        const int kb = k0 + half * 16;
        v16h b[8];
#pragma unroll
        for (int nt = 0; nt < 8; ++nt)
            b[nt] = *(const v16h*)(W + (size_t)(nt * 16 + l16) * C_ + kb);

        // A fragment: x rows, fp32 -> f16.
        const int ka = k0 + half * 16;
        v4f c0 = *(const v4f*)(xrow + ka);
        v4f c1 = *(const v4f*)(xrow + ka + 4);
        v4f c2 = *(const v4f*)(xrow + ka + 8);
        v4f c3 = *(const v4f*)(xrow + ka + 12);
        v16h a;
#pragma unroll
        for (int i = 0; i < 4; ++i) {
            a[i]      = (_Float16)c0[i];
            a[i + 4]  = (_Float16)c1[i];
            a[i + 8]  = (_Float16)c2[i];
            a[i + 12] = (_Float16)c3[i];
        }

#pragma unroll
        for (int nt = 0; nt < 8; ++nt)
            acc[nt] = wmma_f16(a, b[nt], acc[nt]);
    }

    if (which == 2) {
        const int b = m0 / T_;
        const int t = m0 % T_;
        _Float16* vb = vt + (size_t)b * HS_ * T_;
#pragma unroll
        for (int nt = 0; nt < 8; ++nt) {
            const int d = nt * 16 + l16;
#pragma unroll
            for (int r = 0; r < 8; ++r)
                vb[(size_t)d * T_ + (t + r + half * 8)] = (_Float16)acc[nt][r];
        }
    } else {
        _Float16* o = (which == 0) ? qh : kh;
#pragma unroll
        for (int nt = 0; nt < 8; ++nt)
#pragma unroll
            for (int r = 0; r < 8; ++r)
                o[(size_t)(m0 + r + half * 8) * HS_ + nt * 16 + l16] = (_Float16)acc[nt][r];
    }
}

// ---------------------------------------------------------------------------
// Kernel 2: split-K causal flash attention. grid = (BT/16, n_seg).
// Wave (tile, seg) processes keys [seg*seglen, min(kend, (seg+1)*seglen)) and
// writes UNNORMALIZED partial O plus per-row (m, l). Segment boundaries are
// multiples of 512 (tile-aligned), so waves are fully active or skipped.
// ---------------------------------------------------------------------------
__global__ __launch_bounds__(32) void attn_kernel(const _Float16* __restrict__ qh,
                                                  const _Float16* __restrict__ kh,
                                                  const _Float16* __restrict__ vt,
                                                  float* __restrict__ Opart,
                                                  float* __restrict__ ml,
                                                  int seglen) {
    __shared__ _Float16 pbuf[16 * 32];             // P tile: [row][key_local]

    const int tile = blockIdx.x;                   // 0..BT/16-1
    const int seg  = blockIdx.y;
    const int b    = tile >> 8;                    // 256 q-tiles per batch
    const int q0   = (tile & 255) * 16;            // query tile base in batch
    const int kend = q0 + 16;                      // causal: keys < kend
    const int j_start = seg * seglen;
    if (j_start >= kend) return;                   // no keys for this segment
    const int j_end = (kend < j_start + seglen) ? kend : (j_start + seglen);

    const int lane = threadIdx.x & 31;
    const int half = lane >> 4;
    const int l16  = lane & 15;
    const float scale = 0.08838834764831845f;      // 1/sqrt(128)

    const _Float16* qb = qh + (size_t)b * T_ * HS_;
    const _Float16* kb = kh + (size_t)b * T_ * HS_;
    const _Float16* vb = vt + (size_t)b * HS_ * T_;

    // preload Q A-fragments: 4 chunks of K=32 over head dim.
    v16h aq[4];
    {
        const _Float16* qrow = qb + (size_t)(q0 + l16) * HS_;
#pragma unroll
        for (int c = 0; c < 4; ++c) {
            const int d0 = c * 32 + half * 16;
            v8h lo = *(const v8h*)(qrow + d0);
            v8h hi = *(const v8h*)(qrow + d0 + 8);
#pragma unroll
            for (int i = 0; i < 8; ++i) { aq[c][i] = lo[i]; aq[c][i + 8] = hi[i]; }
        }
    }

    v8f o[8];
#pragma unroll
    for (int i = 0; i < 8; ++i) o[i] = (v8f)(0.0f);
    float m_r[8], l_r[8];
#pragma unroll
    for (int r = 0; r < 8; ++r) { m_r[r] = -1e30f; l_r[r] = 0.0f; }

    for (int j0 = j_start; j0 < j_end; j0 += 32) {
        // all 8 K fragments up front (one big load clause).
        v16h bk[8];
#pragma unroll
        for (int jt = 0; jt < 2; ++jt) {
            const int kcol = j0 + jt * 16 + l16;
#pragma unroll
            for (int c = 0; c < 4; ++c)
                bk[jt * 4 + c] =
                    *(const v16h*)(kb + (size_t)kcol * HS_ + c * 32 + half * 16);
        }

        // S = Q K^T for two 16-key subtiles.
        v8f s[2];
#pragma unroll
        for (int jt = 0; jt < 2; ++jt) {
            v8f acc = (v8f)(0.0f);
#pragma unroll
            for (int c = 0; c < 4; ++c)
                acc = wmma_f16(aq[c], bk[jt * 4 + c], acc);
            s[jt] = acc;
        }

        // issue V fragment loads now; softmax hides their latency.
        v16h bv[8];
#pragma unroll
        for (int dt = 0; dt < 8; ++dt)
            bv[dt] = *(const v16h*)(vb + (size_t)(dt * 16 + l16) * T_ + j0 + half * 16);

        // online softmax. Row m = r + half*8 lives in 16 lanes of one half.
#pragma unroll
        for (int r = 0; r < 8; ++r) {
            const int qrow = q0 + r + half * 8;
            float s0 = s[0][r] * scale;
            float s1 = s[1][r] * scale;
            if (j0 + l16 > qrow)      s0 = -1e30f;  // causal mask
            if (j0 + 16 + l16 > qrow) s1 = -1e30f;

            float mx = fmaxf(s0, s1);
#pragma unroll
            for (int off = 1; off < 16; off <<= 1)
                mx = fmaxf(mx, __shfl_xor(mx, off));
            const float mnew = fmaxf(m_r[r], mx);

            const float p0 = __expf(s0 - mnew);
            const float p1 = __expf(s1 - mnew);
            float rs = p0 + p1;
#pragma unroll
            for (int off = 1; off < 16; off <<= 1)
                rs += __shfl_xor(rs, off);

            const float alpha = __expf(m_r[r] - mnew);
            l_r[r] = l_r[r] * alpha + rs;
            m_r[r] = mnew;
#pragma unroll
            for (int dt = 0; dt < 8; ++dt) o[dt][r] *= alpha;

            pbuf[(r + half * 8) * 32 + l16]      = (_Float16)p0;
            pbuf[(r + half * 8) * 32 + 16 + l16] = (_Float16)p1;
        }
        asm volatile("s_wait_dscnt 0" ::: "memory");   // cross-lane LDS transpose

        // P A-fragment (16x32): row = l16, K chunks by lane half.
        v16h ap;
        {
            const _Float16* pr = &pbuf[l16 * 32 + half * 16];
            v8h lo = *(const v8h*)(pr);
            v8h hi = *(const v8h*)(pr + 8);
#pragma unroll
            for (int i = 0; i < 8; ++i) { ap[i] = lo[i]; ap[i + 8] = hi[i]; }
        }

        // O += P V (V fragments already in flight / landed).
#pragma unroll
        for (int dt = 0; dt < 8; ++dt)
            o[dt] = wmma_f16(ap, bv[dt], o[dt]);
    }

    // epilogue: store unnormalized partial O + per-row (m, l) for this segment.
    float* Oseg = Opart + (size_t)seg * BT_ * HS_;
#pragma unroll
    for (int dt = 0; dt < 8; ++dt) {
#pragma unroll
        for (int r = 0; r < 8; ++r) {
            const int row = q0 + r + half * 8;
            Oseg[((size_t)b * T_ + row) * HS_ + dt * 16 + l16] = o[dt][r];
        }
    }
    if (l16 == 0) {
        float* mlseg = ml + (size_t)seg * BT_ * 2;
#pragma unroll
        for (int r = 0; r < 8; ++r) {
            const int row = q0 + r + half * 8;
            mlseg[((size_t)b * T_ + row) * 2 + 0] = m_r[r];
            mlseg[((size_t)b * T_ + row) * 2 + 1] = l_r[r];
        }
    }
}

// ---------------------------------------------------------------------------
// Kernel 3: combine split-K partials: out = sum_s w_s * O_s / sum_s w_s * l_s,
// w_s = exp(m_s - max_s m_s). Only segments with s*seglen <= row%T are valid.
// ---------------------------------------------------------------------------
__global__ __launch_bounds__(256) void combine_kernel(const float* __restrict__ Opart,
                                                      const float* __restrict__ ml,
                                                      float* __restrict__ out,
                                                      int n_seg, int seglen) {
    const int idx = blockIdx.x * 256 + threadIdx.x;   // over BT*HS
    const int row = idx >> 7;                          // global row
    const int rib = row & (T_ - 1);                    // row within batch
    int nv = rib / seglen + 1;
    if (nv > n_seg) nv = n_seg;

    float M = -1e30f;
    for (int s = 0; s < nv; ++s)
        M = fmaxf(M, ml[((size_t)s * BT_ + row) * 2 + 0]);

    float acc = 0.0f, lt = 0.0f;
    for (int s = 0; s < nv; ++s) {
        const float w = __expf(ml[((size_t)s * BT_ + row) * 2 + 0] - M);
        lt  += ml[((size_t)s * BT_ + row) * 2 + 1] * w;
        acc += Opart[(size_t)s * BT_ * HS_ + idx] * w;
    }
    out[idx] = acc / lt;
}

// ---------------------------------------------------------------------------
extern "C" void kernel_launch(void* const* d_in, const int* in_sizes, int n_in,
                              void* d_out, int out_size, void* d_ws, size_t ws_size,
                              hipStream_t stream) {
    const float* x  = (const float*)d_in[0];
    const float* Wq = (const float*)d_in[1];
    const float* Wk = (const float*)d_in[2];
    const float* Wv = (const float*)d_in[3];
    float* out = (float*)d_out;

    // ws layout (f16): Wt[3][128][2048] | qh[BT][128] | kh[BT][128] | vt[B][128][T]
    _Float16* wt = (_Float16*)d_ws;
    _Float16* qh = wt + (size_t)3 * HS_ * C_;
    _Float16* kh = qh + (size_t)BT_ * HS_;
    _Float16* vt = kh + (size_t)BT_ * HS_;

    // fp32 split-K scratch after the f16 region (256B aligned).
    size_t base = ((size_t)(3 * HS_ * C_ + 3 * BT_ * HS_) * 2 + 255) & ~(size_t)255;
    const size_t per_seg = (size_t)BT_ * HS_ * 4 + (size_t)BT_ * 2 * 4;

    int n_seg = 1;                                  // deterministic w.r.t. ws_size
    for (int cand = 8; cand > 1; cand >>= 1)
        if (base + (size_t)cand * per_seg <= ws_size) { n_seg = cand; break; }
    const int seglen = T_ / n_seg;                  // multiple of 512 -> tile-aligned

    float* Opart = (float*)((char*)d_ws + base);
    float* ml    = Opart + (size_t)n_seg * BT_ * HS_;

    // 1) weight transpose + f16 convert
    wconv_kernel<<<dim3((C_ * HS_) / 256, 3), 256, 0, stream>>>(Wq, Wk, Wv, wt);

    // 2) QKV projection (one wave per 16x128 tile; grid.y selects q/k/v)
    qkv_kernel<<<dim3(BT_ / 16, 3), 32, 0, stream>>>(x, wt, qh, kh, vt);

    // 3) split-K causal flash attention (one wave per (q-tile, key-segment))
    attn_kernel<<<dim3(BT_ / 16, n_seg), 32, 0, stream>>>(qh, kh, vt, Opart, ml, seglen);

    // 4) combine segment partials -> fp32 output
    combine_kernel<<<(BT_ * HS_) / 256, 256, 0, stream>>>(Opart, ml, out, n_seg, seglen);
}
